// Decoder_21500606284155
// MI455X (gfx1250) — matrix operand order repaired and checked by reading, not compiled, not run on hardware
//
#include <hip/hip_runtime.h>
#include <hip/hip_bf16.h>
#include <stdint.h>

// ---------------------------------------------------------------------------
// Types
// ---------------------------------------------------------------------------
typedef unsigned int  u32;
typedef unsigned short u16;
typedef __attribute__((ext_vector_type(16))) __bf16 v16bf;
typedef __attribute__((ext_vector_type(8)))  float  v8f;

union Pack {
    uint4 u[2];
    v16bf v;
};

__device__ __forceinline__ u16 f2bf(float f) {
    u32 u = __float_as_uint(f);
    u32 r = u + 0x7FFFu + ((u >> 16) & 1u);   // round-to-nearest-even
    return (u16)(r >> 16);
}
__device__ __forceinline__ float bf2f(u16 h) {
    return __uint_as_float(((u32)h) << 16);
}
__device__ __forceinline__ float sigm(float x) {
    return 1.0f / (1.0f + __expf(-x));
}

__device__ __forceinline__ v8f wmma_bf16(v16bf a, v16bf b, v8f c) {
    // D(f32 16x16) = A(bf16 16x32) * B(bf16 32x16) + C
    return __builtin_amdgcn_wmma_f32_16x16x32_bf16(
        false, a, false, b, (short)0, c, false, false);
}

// ---------------------------------------------------------------------------
// CDNA5 async global->LDS copy (ASYNCcnt-tracked, no staging VGPRs).
// LDS generic addresses carry the LDS byte offset in their low 32 bits.
// ---------------------------------------------------------------------------
__device__ __forceinline__ void async_b128(void* lds, const void* gptr) {
    u32 loff = (u32)(uintptr_t)lds;
    unsigned long long ga = (unsigned long long)(uintptr_t)gptr;
    asm volatile("global_load_async_to_lds_b128 %0, %1, off"
                 :: "v"(loff), "v"(ga) : "memory");
}
__device__ __forceinline__ void wait_async_3() {
    asm volatile("s_wait_asynccnt 0x3" ::: "memory");
}
__device__ __forceinline__ void wait_async_0() {
    asm volatile("s_wait_asynccnt 0x0" ::: "memory");
}

// ---------------------------------------------------------------------------
// fp32 -> bf16 conversion with optional zero padding of trailing columns
// ---------------------------------------------------------------------------
__global__ void cvt_kernel(const float* __restrict__ src, u16* __restrict__ dst,
                           int rows, int scols, int dcols) {
    size_t i = (size_t)blockIdx.x * blockDim.x + threadIdx.x;
    size_t total = (size_t)rows * dcols;
    if (i >= total) return;
    int r = (int)(i / dcols);
    int c = (int)(i % dcols);
    dst[i] = (c < scols) ? f2bf(src[(size_t)r * scols + c]) : (u16)0;
}

__global__ void addbias_kernel(const float* a, const float* b, float* out, int n) {
    int i = blockIdx.x * blockDim.x + threadIdx.x;
    if (i < n) out[i] = a[i] + b[i];
}

// ---------------------------------------------------------------------------
// bf16 GEMM:  C[M,N] = A[M,K] * B[N,K]^T + bias  (optional tanh)
//   A: bf16 MxK row-major (M padded to 64), B: bf16 NxK row-major
//   M % 64 == 0, N % 128 == 0, K % 32 == 0; stores guarded by m < mvalid
//   remap: C row m -> (m/29)*30 + m%29 + 1   (decoder output layout)
// Block = 256 thr = 8 waves; block tile 64x128; wave tile 32x32 (2x2 WMMA).
// Double-buffered LDS filled by async global->LDS copies; last K-block is
// peeled so the steady-state loop is branch-free:
//   stage(next) ; s_wait_asynccnt 3 ; barrier ; 8x ds_load + 4x wmma ; barrier
// ---------------------------------------------------------------------------
#define BM 64
#define BN 128
#define BK 32

__global__ __launch_bounds__(256) void gemm_bf16_kernel(
    const u16* __restrict__ A, const u16* __restrict__ B,
    const float* __restrict__ bias, float* __restrict__ C,
    int M, int N, int K, int act, int remap, int mvalid)
{
    __shared__ __align__(16) u16 As[2][BM][BK];   // 2 x 4 KB
    __shared__ __align__(16) u16 Bs[2][BN][BK];   // 2 x 8 KB

    const int tid  = threadIdx.x;
    const int wave = tid >> 5;
    const int lane = tid & 31;
    const int half = lane >> 4;
    const int l16  = lane & 15;
    const int m0 = blockIdx.y * BM;
    const int n0 = blockIdx.x * BN;
    const int wm = wave & 1;    // M half  (rows wm*32 .. +31)
    const int wn = wave >> 1;   // N quad  (cols wn*32 .. +31)

    // staging coordinates: A 64x32 (1 x b128/thr), B 128x32 (2 x b128/thr)
    const int ar = tid >> 2, ac = (tid & 3) * 8;
    const int br = tid >> 1, bc = (tid & 1) * 16;

    // hoisted per-thread global base pointers (K-offset varies per stage)
    const u16* __restrict__ Ag = A + (size_t)(m0 + ar) * K + ac;
    const u16* __restrict__ Bg = B + (size_t)(n0 + br) * K + bc;

    const int nk = K / BK;

    v8f acc[2][2];
#pragma unroll
    for (int mt = 0; mt < 2; ++mt)
#pragma unroll
        for (int nt = 0; nt < 2; ++nt) acc[mt][nt] = (v8f){};

    auto stage = [&](int buf, int k0) {
        async_b128(&As[buf][ar][ac],     Ag + k0);
        async_b128(&Bs[buf][br][bc],     Bg + k0);
        async_b128(&Bs[buf][br][bc + 8], Bg + k0 + 8);
    };
    auto compute = [&](int buf) {
        // A fragment (ISA 16-bit A 16x32 layout):
        //   lanes 0-15: row l16, K {0..7,16..23}; lanes 16-31: K {8..15,24..31}
        Pack pa[2], pb[2];
#pragma unroll
        for (int mt = 0; mt < 2; ++mt) {
            pa[mt].u[0] = *(const uint4*)&As[buf][wm * 32 + mt * 16 + l16][half * 8];
            pa[mt].u[1] = *(const uint4*)&As[buf][wm * 32 + mt * 16 + l16][16 + half * 8];
        }
        // B fragment: lanes 0-15 col l16 K {0..7,16..23}; lanes 16-31 col l16 K {8..15,24..31}
#pragma unroll
        for (int nt = 0; nt < 2; ++nt) {
            pb[nt].u[0] = *(const uint4*)&Bs[buf][wn * 32 + nt * 16 + l16][half * 8];
            pb[nt].u[1] = *(const uint4*)&Bs[buf][wn * 32 + nt * 16 + l16][16 + half * 8];
        }
#pragma unroll
        for (int mt = 0; mt < 2; ++mt)
#pragma unroll
            for (int nt = 0; nt < 2; ++nt)
                acc[mt][nt] = wmma_bf16(pa[mt].v, pb[nt].v, acc[mt][nt]);
    };

    // prologue: stage K-block 0 into buffer 0
    stage(0, 0);

    // steady state: branch-free
    for (int kb = 0; kb < nk - 1; ++kb) {
        const int cur = kb & 1;
        stage(cur ^ 1, (kb + 1) * BK);
        wait_async_3();          // previous stage (buffer cur) complete
        __syncthreads();         // ... and visible to all waves
        compute(cur);
        __syncthreads();         // compute done before buffer is rewritten
    }
    // peeled final K-block
    {
        const int cur = (nk - 1) & 1;
        wait_async_0();
        __syncthreads();
        compute(cur);
    }

    // Epilogue. C/D layout: VGPR r -> M = half*8 + r, N = l16.
#pragma unroll
    for (int mt = 0; mt < 2; ++mt) {
#pragma unroll
        for (int nt = 0; nt < 2; ++nt) {
            const int n = n0 + wn * 32 + nt * 16 + l16;
            const float bb = bias ? bias[n] : 0.0f;
#pragma unroll
            for (int r = 0; r < 8; ++r) {
                const int m = m0 + wm * 32 + mt * 16 + half * 8 + r;
                if (m >= mvalid) continue;
                float v = acc[mt][nt][r] + bb;
                if (act == 1) v = tanhf(v);
                size_t row = remap ? ((size_t)(m / 29) * 30 + (m % 29) + 1)
                                   : (size_t)m;
                C[row * (size_t)N + n] = v;
            }
        }
    }
}

// ---------------------------------------------------------------------------
// Persistent bidirectional LSTM scan. grid.x = 2 (dir), block = 1024 (32 waves).
// Each wave owns 2 "gate quads" (fixed 16-row x 16-H-col tiles whose 4
// accumulators are i/f/g/o), cell state in registers, h in LDS as bf16.
// ---------------------------------------------------------------------------
__global__ __launch_bounds__(1024) void lstm_scan_kernel(
    const float* __restrict__ pre_f, const float* __restrict__ pre_b,
    const u16* __restrict__ Whh_f, const u16* __restrict__ Whh_b,
    u16* __restrict__ encbf, u16* __restrict__ hTbf)
{
    const int dir = blockIdx.x;
    const float* __restrict__ pre = dir ? pre_b : pre_f;
    const u16*   __restrict__ Whh = dir ? Whh_b : Whh_f;

    __shared__ __align__(16) u16 hbuf[32][512];   // 32 KB

    const int tid = threadIdx.x;
    for (int i = tid; i < 32 * 512; i += 1024) ((u16*)hbuf)[i] = 0;
    __syncthreads();

    const int wave = tid >> 5;
    const int lane = tid & 31;
    const int half = lane >> 4;
    const int l16  = lane & 15;

    float cst[2][8];
#pragma unroll
    for (int q = 0; q < 2; ++q)
#pragma unroll
        for (int r = 0; r < 8; ++r) cst[q][r] = 0.0f;

    for (int step = 0; step < 100; ++step) {
        const int t = dir ? (99 - step) : step;

        v8f acc[2][4];
#pragma unroll
        for (int qi = 0; qi < 2; ++qi) {
            const int q  = wave * 2 + qi;   // 0..63
            const int mt = q & 1;
            const int nq = q >> 1;          // 0..31 : 16-col block of H
#pragma unroll
            for (int g = 0; g < 4; ++g) acc[qi][g] = (v8f){};
            for (int k0 = 0; k0 < 512; k0 += 32) {
                Pack pa;
                pa.u[0] = *(const uint4*)&hbuf[mt * 16 + l16][k0 + half * 8];
                pa.u[1] = *(const uint4*)&hbuf[mt * 16 + l16][k0 + 16 + half * 8];
#pragma unroll
                for (int g = 0; g < 4; ++g) {
                    const int n = g * 512 + nq * 16 + l16;
                    if (k0 + 32 < 512)   // keep next K-slice warm (global_prefetch)
                        __builtin_prefetch(&Whh[(size_t)n * 512 + k0 + 32], 0, 0);
                    Pack pb;
                    pb.u[0] = *(const uint4*)&Whh[(size_t)n * 512 + k0 + half * 8];
                    pb.u[1] = *(const uint4*)&Whh[(size_t)n * 512 + k0 + 16 + half * 8];
                    acc[qi][g] = wmma_bf16(pa.v, pb.v, acc[qi][g]);
                }
            }
        }
        __syncthreads();   // all hbuf reads done before rewrite

#pragma unroll
        for (int qi = 0; qi < 2; ++qi) {
            const int q  = wave * 2 + qi;
            const int mt = q & 1;
            const int nq = q >> 1;
            const int n  = nq * 16 + l16;               // H column
#pragma unroll
            for (int r = 0; r < 8; ++r) {
                const int b = mt * 16 + half * 8 + r;    // batch row
                const size_t prow = ((size_t)b * 100 + t) * 2048;
                float gi_ = acc[qi][0][r] + pre[prow + 0 * 512 + n];
                float gf  = acc[qi][1][r] + pre[prow + 1 * 512 + n];
                float gg  = acc[qi][2][r] + pre[prow + 2 * 512 + n];
                float go  = acc[qi][3][r] + pre[prow + 3 * 512 + n];
                float c = sigm(gf) * cst[qi][r] + sigm(gi_) * tanhf(gg);
                cst[qi][r] = c;
                float h = sigm(go) * tanhf(c);
                u16 hb = f2bf(h);
                hbuf[b][n] = hb;
                encbf[((size_t)b * 100 + t) * 1024 + dir * 512 + n] = hb;
                if (step == 99) hTbf[(size_t)b * 1024 + dir * 512 + n] = hb;
            }
        }
        __syncthreads();
    }
}

// ---------------------------------------------------------------------------
// Attention concat: catbf[b*S+s][0..1023]=encbf, [1024..1535]=stbf[b]
// ---------------------------------------------------------------------------
__global__ void cat_attn_kernel(const u16* __restrict__ encbf,
                                const u16* __restrict__ stbf,
                                u16* __restrict__ catbf) {
    size_t i = (size_t)blockIdx.x * blockDim.x + threadIdx.x;
    size_t total = (size_t)3200 * 1536;
    if (i >= total) return;
    int row = (int)(i / 1536);
    int c   = (int)(i % 1536);
    int b   = row / 100;
    catbf[i] = (c < 1024) ? encbf[(size_t)row * 1024 + c]
                          : stbf[(size_t)b * 512 + (c - 1024)];
}

// ---------------------------------------------------------------------------
// scores[b,s] = energy[b,s,:] . attn_w2 + attn_b2   (one wave per row)
// ---------------------------------------------------------------------------
__global__ void scores_kernel(const float* __restrict__ energy,
                              const float* __restrict__ w2,
                              const float* __restrict__ b2,
                              float* __restrict__ scores) {
    int row  = blockIdx.x * 4 + (threadIdx.x >> 5);   // 3200 rows
    int lane = threadIdx.x & 31;
    const float* e = energy + (size_t)row * 512;
    float s = 0.0f;
    for (int k = lane; k < 512; k += 32) s += e[k] * w2[k];
#pragma unroll
    for (int off = 16; off; off >>= 1) s += __shfl_down(s, off, 32);
    if (lane == 0) scores[row] = s + b2[0];
}

// ---------------------------------------------------------------------------
// softmax over S=100, one block per batch row
// ---------------------------------------------------------------------------
__global__ void softmax_kernel(const float* __restrict__ scores,
                               float* __restrict__ attn) {
    __shared__ float red[4];
    const int b = blockIdx.x, tid = threadIdx.x;
    const int lane = tid & 31, wv = tid >> 5;
    float m = -1e30f;
    for (int s = tid; s < 100; s += 128) m = fmaxf(m, scores[b * 100 + s]);
#pragma unroll
    for (int off = 16; off; off >>= 1) m = fmaxf(m, __shfl_xor(m, off, 32));
    if (lane == 0) red[wv] = m;
    __syncthreads();
    m = fmaxf(fmaxf(red[0], red[1]), fmaxf(red[2], red[3]));
    __syncthreads();
    float sum = 0.0f;
    for (int s = tid; s < 100; s += 128) sum += __expf(scores[b * 100 + s] - m);
#pragma unroll
    for (int off = 16; off; off >>= 1) sum += __shfl_xor(sum, off, 32);
    if (lane == 0) red[wv] = sum;
    __syncthreads();
    sum = red[0] + red[1] + red[2] + red[3];
    for (int s = tid; s < 100; s += 128)
        attn[b * 100 + s] = __expf(scores[b * 100 + s] - m) / sum;
}

// ---------------------------------------------------------------------------
// attented[b,d] = sum_s attn[b,s] * enc[b,s,d];  also writes bf16 copy
// ---------------------------------------------------------------------------
__global__ void attented_kernel(const float* __restrict__ attn,
                                const u16* __restrict__ encbf,
                                float* __restrict__ attented,
                                u16* __restrict__ attbf) {
    int b = blockIdx.x;
    int d = blockIdx.y * 256 + threadIdx.x;   // 0..1023
    float s = 0.0f;
    for (int t = 0; t < 100; ++t)
        s += attn[b * 100 + t] * bf2f(encbf[((size_t)b * 100 + t) * 1024 + d]);
    attented[(size_t)b * 1024 + d] = s;
    attbf[(size_t)b * 1024 + d]   = f2bf(s);
}

// ---------------------------------------------------------------------------
// GRU input: rnnin[960][1344] = [attented(1024) | emb_eff[tok](300) | pad(20)]
// (rows >= 928 are padding; never consumed downstream)
// ---------------------------------------------------------------------------
__global__ void rnnin_kernel(const u16* __restrict__ attbf,
                             const float* __restrict__ emb,
                             const int* __restrict__ ans,
                             u16* __restrict__ rnnin) {
    size_t i = (size_t)blockIdx.x * blockDim.x + threadIdx.x;
    size_t total = (size_t)928 * 1344;
    if (i >= total) return;
    int row = (int)(i / 1344);
    int c   = (int)(i % 1344);
    int b = row / 29, t = row % 29;
    u16 v;
    if (c < 1024) {
        v = attbf[(size_t)b * 1024 + c];
    } else if (c < 1324) {
        int tok = ans[b * 30 + t];                 // true_answer[:, :-1]
        v = (tok == 0) ? (u16)0 : f2bf(emb[(size_t)tok * 300 + (c - 1024)]);
    } else {
        v = 0;
    }
    rnnin[i] = v;
}

// ---------------------------------------------------------------------------
// GRU gate combine -> h_new written as bf16 into decin cols [0,512)
// ---------------------------------------------------------------------------
__global__ void gru_combine_kernel(const float* __restrict__ gi,
                                   const float* __restrict__ gh,
                                   const float* __restrict__ init_state,
                                   u16* __restrict__ decin) {
    size_t i = (size_t)blockIdx.x * blockDim.x + threadIdx.x;
    size_t total = (size_t)928 * 512;
    if (i >= total) return;
    int row = (int)(i / 512);
    int n   = (int)(i % 512);
    int b = row / 29;
    float r  = sigm(gi[(size_t)row * 1536 + n]        + gh[(size_t)b * 1536 + n]);
    float z  = sigm(gi[(size_t)row * 1536 + 512 + n]  + gh[(size_t)b * 1536 + 512 + n]);
    float nn = tanhf(gi[(size_t)row * 1536 + 1024 + n] +
                     r * gh[(size_t)b * 1536 + 1024 + n]);
    float h = (1.0f - z) * nn + z * init_state[(size_t)b * 512 + n];
    decin[(size_t)row * 1856 + n] = f2bf(h);
}

// ---------------------------------------------------------------------------
// decin cols [512,1856): [prev(300) | attented(1024) | pad(20)]
// ---------------------------------------------------------------------------
__global__ void decin_rest_kernel(const u16* __restrict__ rnnin,
                                  const u16* __restrict__ attbf,
                                  u16* __restrict__ decin) {
    size_t i = (size_t)blockIdx.x * blockDim.x + threadIdx.x;
    size_t total = (size_t)928 * 1344;
    if (i >= total) return;
    int row = (int)(i / 1344);
    int c   = 512 + (int)(i % 1344);
    int b = row / 29;
    u16 v;
    if (c < 812)        v = rnnin[(size_t)row * 1344 + 1024 + (c - 512)]; // prev
    else if (c < 1836)  v = attbf[(size_t)b * 1024 + (c - 812)];          // att
    else                v = 0;                                             // pad
    decin[(size_t)row * 1856 + c] = v;
}

__global__ void zero_t0_kernel(float* __restrict__ out) {
    size_t i = (size_t)blockIdx.x * blockDim.x + threadIdx.x;
    if (i >= (size_t)32 * 32000) return;
    size_t b = i / 32000, n = i % 32000;
    out[(b * 30) * 32000 + n] = 0.0f;
}

// ---------------------------------------------------------------------------
// Host launch
// ---------------------------------------------------------------------------
static inline char* carve(char*& p, size_t bytes) {
    char* r = p;
    p += (bytes + 255) & ~(size_t)255;
    return r;
}
static inline unsigned grid1d(size_t n, unsigned blk) {
    return (unsigned)((n + blk - 1) / blk);
}

extern "C" void kernel_launch(void* const* d_in, const int* in_sizes, int n_in,
                              void* d_out, int out_size, void* d_ws, size_t ws_size,
                              hipStream_t stream) {
    (void)in_sizes; (void)n_in; (void)out_size; (void)ws_size;

    const float* span    = (const float*)d_in[0];
    const int*   ans     = (const int*)  d_in[1];
    const float* W_ih_f  = (const float*)d_in[2];
    const float* W_hh_f  = (const float*)d_in[3];
    const float* b_ih_f  = (const float*)d_in[4];
    const float* b_hh_f  = (const float*)d_in[5];
    const float* W_ih_b  = (const float*)d_in[6];
    const float* W_hh_b  = (const float*)d_in[7];
    const float* b_ih_b  = (const float*)d_in[8];
    const float* b_hh_b  = (const float*)d_in[9];
    const float* fc_w    = (const float*)d_in[10];
    const float* fc_b    = (const float*)d_in[11];
    const float* emb     = (const float*)d_in[12];
    const float* attn_w1 = (const float*)d_in[13];
    const float* attn_b1 = (const float*)d_in[14];
    const float* attn_w2 = (const float*)d_in[15];
    const float* attn_b2 = (const float*)d_in[16];
    const float* gru_Wih = (const float*)d_in[17];
    const float* gru_Whh = (const float*)d_in[18];
    const float* gru_bih = (const float*)d_in[19];
    const float* gru_bhh = (const float*)d_in[20];
    const float* out_w   = (const float*)d_in[21];
    const float* out_b   = (const float*)d_in[22];
    float* out = (float*)d_out;

    // ---- workspace layout (M dims padded to 64-row tiles) -----------------
    char* p = (char*)d_ws;
    u16*  spanbf    = (u16*)  carve(p, (size_t)3200 * 512 * 2);
    u16*  Wihf_bf   = (u16*)  carve(p, (size_t)2048 * 512 * 2);
    u16*  Wihb_bf   = (u16*)  carve(p, (size_t)2048 * 512 * 2);
    u16*  Whhf_bf   = (u16*)  carve(p, (size_t)2048 * 512 * 2);
    u16*  Whhb_bf   = (u16*)  carve(p, (size_t)2048 * 512 * 2);
    u16*  fcw_bf    = (u16*)  carve(p, (size_t)512 * 1024 * 2);
    u16*  attnw1_bf = (u16*)  carve(p, (size_t)512 * 1536 * 2);
    u16*  gruWhh_bf = (u16*)  carve(p, (size_t)1536 * 512 * 2);
    u16*  gruWih_bf = (u16*)  carve(p, (size_t)1536 * 1344 * 2);
    u16*  outw_bf   = (u16*)  carve(p, (size_t)32000 * 1856 * 2);
    float* bias_f   = (float*)carve(p, (size_t)2048 * 4);
    float* bias_b   = (float*)carve(p, (size_t)2048 * 4);
    float* pre_f    = (float*)carve(p, (size_t)3200 * 2048 * 4);
    float* pre_b    = (float*)carve(p, (size_t)3200 * 2048 * 4);
    u16*  encbf     = (u16*)  carve(p, (size_t)3200 * 1024 * 2);
    u16*  hTbf      = (u16*)  carve(p, (size_t)64 * 1024 * 2);    // 64-row pad
    float* init_st  = (float*)carve(p, (size_t)32 * 512 * 4);
    u16*  stbf      = (u16*)  carve(p, (size_t)64 * 512 * 2);     // 64-row pad
    u16*  catbf     = (u16*)  carve(p, (size_t)3200 * 1536 * 2);
    float* energy   = (float*)carve(p, (size_t)3200 * 512 * 4);
    float* scores   = (float*)carve(p, (size_t)3200 * 4);
    float* attnp    = (float*)carve(p, (size_t)3200 * 4);
    float* attented = (float*)carve(p, (size_t)32 * 1024 * 4);
    u16*  attbf     = (u16*)  carve(p, (size_t)32 * 1024 * 2);
    u16*  rnnin     = (u16*)  carve(p, (size_t)960 * 1344 * 2);   // 960-row pad
    float* gi       = (float*)carve(p, (size_t)960 * 1536 * 4);   // 960-row pad
    float* gh       = (float*)carve(p, (size_t)32 * 1536 * 4);
    u16*  decin     = (u16*)  carve(p, (size_t)960 * 1856 * 2);   // 960-row pad

    const unsigned CB = 256;
    // ---- fp32 -> bf16 conversions ----------------------------------------
    cvt_kernel<<<grid1d((size_t)3200 * 512, CB), CB, 0, stream>>>(span, spanbf, 3200, 512, 512);
    cvt_kernel<<<grid1d((size_t)2048 * 512, CB), CB, 0, stream>>>(W_ih_f, Wihf_bf, 2048, 512, 512);
    cvt_kernel<<<grid1d((size_t)2048 * 512, CB), CB, 0, stream>>>(W_ih_b, Wihb_bf, 2048, 512, 512);
    cvt_kernel<<<grid1d((size_t)2048 * 512, CB), CB, 0, stream>>>(W_hh_f, Whhf_bf, 2048, 512, 512);
    cvt_kernel<<<grid1d((size_t)2048 * 512, CB), CB, 0, stream>>>(W_hh_b, Whhb_bf, 2048, 512, 512);
    cvt_kernel<<<grid1d((size_t)512 * 1024, CB), CB, 0, stream>>>(fc_w, fcw_bf, 512, 1024, 1024);
    cvt_kernel<<<grid1d((size_t)512 * 1536, CB), CB, 0, stream>>>(attn_w1, attnw1_bf, 512, 1536, 1536);
    cvt_kernel<<<grid1d((size_t)1536 * 512, CB), CB, 0, stream>>>(gru_Whh, gruWhh_bf, 1536, 512, 512);
    cvt_kernel<<<grid1d((size_t)1536 * 1344, CB), CB, 0, stream>>>(gru_Wih, gruWih_bf, 1536, 1324, 1344);
    cvt_kernel<<<grid1d((size_t)32000 * 1856, CB), CB, 0, stream>>>(out_w, outw_bf, 32000, 1836, 1856);

    addbias_kernel<<<grid1d(2048, CB), CB, 0, stream>>>(b_ih_f, b_hh_f, bias_f, 2048);
    addbias_kernel<<<grid1d(2048, CB), CB, 0, stream>>>(b_ih_b, b_hh_b, bias_b, 2048);

    // ---- LSTM input pre-GEMMs: pre = span @ Wih^T + (bih+bhh) ------------
    gemm_bf16_kernel<<<dim3(2048 / BN, 3200 / BM), 256, 0, stream>>>(
        spanbf, Wihf_bf, bias_f, pre_f, 3200, 2048, 512, 0, 0, 3200);
    gemm_bf16_kernel<<<dim3(2048 / BN, 3200 / BM), 256, 0, stream>>>(
        spanbf, Wihb_bf, bias_b, pre_b, 3200, 2048, 512, 0, 0, 3200);

    // ---- persistent LSTM scan (both directions) --------------------------
    lstm_scan_kernel<<<2, 1024, 0, stream>>>(pre_f, pre_b, Whhf_bf, Whhb_bf, encbf, hTbf);

    // ---- init_state = tanh(hidden @ fc_w^T + fc_b) -----------------------
    gemm_bf16_kernel<<<dim3(512 / BN, 1), 256, 0, stream>>>(
        hTbf, fcw_bf, fc_b, init_st, 64, 512, 1024, 1, 0, 32);
    cvt_kernel<<<grid1d((size_t)32 * 512, CB), CB, 0, stream>>>(init_st, stbf, 32, 512, 512);

    // ---- attention -------------------------------------------------------
    cat_attn_kernel<<<grid1d((size_t)3200 * 1536, CB), CB, 0, stream>>>(encbf, stbf, catbf);
    gemm_bf16_kernel<<<dim3(512 / BN, 3200 / BM), 256, 0, stream>>>(
        catbf, attnw1_bf, attn_b1, energy, 3200, 512, 1536, 1, 0, 3200);
    scores_kernel<<<800, 128, 0, stream>>>(energy, attn_w2, attn_b2, scores);
    softmax_kernel<<<32, 128, 0, stream>>>(scores, attnp);
    attented_kernel<<<dim3(32, 4), 256, 0, stream>>>(attnp, encbf, attented, attbf);

    // ---- GRU -------------------------------------------------------------
    rnnin_kernel<<<grid1d((size_t)928 * 1344, CB), CB, 0, stream>>>(attbf, emb, ans, rnnin);
    gemm_bf16_kernel<<<dim3(1536 / BN, 960 / BM), 256, 0, stream>>>(
        rnnin, gruWih_bf, gru_bih, gi, 960, 1536, 1344, 0, 0, 928);
    gemm_bf16_kernel<<<dim3(1536 / BN, 1), 256, 0, stream>>>(
        stbf, gruWhh_bf, gru_bhh, gh, 64, 1536, 512, 0, 0, 32);
    gru_combine_kernel<<<grid1d((size_t)928 * 512, CB), CB, 0, stream>>>(gi, gh, init_st, decin);
    decin_rest_kernel<<<grid1d((size_t)928 * 1344, CB), CB, 0, stream>>>(rnnin, attbf, decin);

    // ---- output projection (dominant GEMM) + t=0 zeros -------------------
    zero_t0_kernel<<<grid1d((size_t)32 * 32000, CB), CB, 0, stream>>>(out);
    gemm_bf16_kernel<<<dim3(32000 / BN, 960 / BM), 256, 0, stream>>>(
        decin, outw_bf, out_b, out, 960, 32000, 1856, 0, 1, 928);
}
